// MPNNv1_75179107549845
// MI455X (gfx1250) — matrix operand order repaired and checked by reading, not compiled, not run on hardware
//
#include <hip/hip_runtime.h>
#include <math.h>

// ---------------------------------------------------------------------------
// MPNN on MI455X (gfx1250).
// - All [*,H]x[H,H] GEMMs on v_wmma_f32_16x16x32_f16 (f32 accumulate).
// - Hidden dim padded to Hp=224 (7 K-chunks of 32); N=100000 and G=64 are
//   exact multiples of 16 -> NO guards anywhere in the GEMM hot path.
// - Weights pre-packed once into the wave32 WMMA B-operand swizzle (f16),
//   activations kept as f16 A-operand copies via dual-store epilogues.
// - One wave owns a 16x224 output strip: A loaded once per K-chunk, 14 (or 28
//   for dual-input GEMMs) consecutive v_wmma per chunk.
// - Edge phase: gate recomputed in-register, scatter-add via f32 atomics.
// ---------------------------------------------------------------------------

typedef __attribute__((ext_vector_type(16))) _Float16 v16h;
typedef __attribute__((ext_vector_type(8)))  _Float16 v8h;
typedef __attribute__((ext_vector_type(8)))  float    v8f;

#define DEVINL __device__ __forceinline__

constexpr int NN = 100000;   // nodes (6250 * 16)
constexpr int NE = 800000;   // edges
constexpr int FN = 128;      // node feature dim (4 K-chunks)
constexpr int FE = 16;       // edge feature dim
constexpr int H  = 200;      // hidden
constexpr int DT = 25;       // per-tower dim
constexpr int NG = 64;       // graphs (4 row tiles)
constexpr int STEPS = 3;

constexpr int Hp  = 224;     // padded hidden = 7*32
constexpr int CT  = Hp / 16; // 14 col tiles
constexpr int KCH = Hp / 32; // 7 K-chunks (K=200 padded)
constexpr int KCF = FN / 32; // 4 K-chunks (K=128)

enum { EPI_RELU = 0, EPI_SIG = 1, EPI_SIGMUL = 2, EPI_GRU = 3, EPI_NONE = 4 };

DEVINL float sigmoidf_(float x) { return 1.0f / (1.0f + __expf(-x)); }

DEVINL v16h make16(v8h lo, v8h hi) {
  union { v16h v; v8h h[2]; } u;
  u.h[0] = lo; u.h[1] = hi;
  return u.v;
}

// ---- Fused GEMM over a 16x224 output strip per wave ------------------------
// out = epi( A1@B1 [+ A2@B2] + biasp ; aux0, aux1 ), stores f32 and/or f16.
// A: f16 [rows, lda] row-major, lda in {Hp, FN}; loads are 16B-aligned.
// B: f16 pre-swizzled: halves index = ((kc*CT + ct)*32 + lane)*16 + j.
template <int EPI, bool DUAL>
__global__ __launch_bounds__(256) void gemm_wmma_kernel(
    const _Float16* __restrict__ A1, const _Float16* __restrict__ B1,
    const _Float16* __restrict__ A2, const _Float16* __restrict__ B2,
    const float* __restrict__ biasp,                       // [Hp], zero padded
    const float* __restrict__ aux0, const float* __restrict__ aux1,  // [*,Hp]
    float* __restrict__ out32, _Float16* __restrict__ out16,
    int rowTiles, int kChunks, int lda) {
  const int lane = threadIdx.x & 31;
  const int tile = blockIdx.x * (blockDim.x >> 5) + (threadIdx.x >> 5);
  if (tile >= rowTiles) return;
  const int mr   = lane & 15;
  const int half = lane >> 4;
  const int row0 = tile * 16;

  v8f zero = {};
  v8f acc[CT];
#pragma unroll
  for (int t = 0; t < CT; ++t) acc[t] = zero;

  // A per-lane: a[0..7] = A[row][k0+half*8 ..], a[8..15] = A[row][k0+16+half*8 ..]
  const _Float16* ap1 = A1 + (size_t)(row0 + mr) * lda + half * 8;
  const _Float16* ap2 = DUAL ? (A2 + (size_t)(row0 + mr) * lda + half * 8) : nullptr;
  const _Float16* bl1 = B1 + (size_t)lane * 16;
  const _Float16* bl2 = DUAL ? (B2 + (size_t)lane * 16) : nullptr;

  for (int kc = 0; kc < kChunks; ++kc) {
    const size_t bofs = (size_t)kc * CT * 512;
    v16h a1 = make16(*(const v8h*)ap1, *(const v8h*)(ap1 + 16));
    __builtin_prefetch(ap1 + 32, 0, 0);   // global_prefetch next K-chunk
#pragma unroll
    for (int ct = 0; ct < CT; ++ct) {
      v16h b1 = *(const v16h*)(bl1 + bofs + (size_t)ct * 512);
      acc[ct] = __builtin_amdgcn_wmma_f32_16x16x32_f16(
          false, a1, false, b1, (short)0, acc[ct], false, false);
    }
    if (DUAL) {
      v16h a2 = make16(*(const v8h*)ap2, *(const v8h*)(ap2 + 16));
#pragma unroll
      for (int ct = 0; ct < CT; ++ct) {
        v16h b2 = *(const v16h*)(bl2 + bofs + (size_t)ct * 512);
        acc[ct] = __builtin_amdgcn_wmma_f32_16x16x32_f16(
            false, a2, false, b2, (short)0, acc[ct], false, false);
      }
    }
    ap1 += 32;
    if (DUAL) ap2 += 32;
  }

  // C/D layout: acc[ct][i] = C[row0 + i + 8*half][ct*16 + mr]
#pragma unroll
  for (int ct = 0; ct < CT; ++ct) {
    const int col = ct * 16 + mr;
    const float bv = biasp[col];
#pragma unroll
    for (int i = 0; i < 8; ++i) {
      const int rr = row0 + i + 8 * half;
      const size_t idx = (size_t)rr * Hp + col;
      float v = acc[ct][i] + bv;
      if (EPI == EPI_RELU)        v = fmaxf(v, 0.0f);
      else if (EPI == EPI_SIG)    v = sigmoidf_(v);
      else if (EPI == EPI_SIGMUL) v = sigmoidf_(v) * aux0[idx];          // r*h
      else if (EPI == EPI_GRU) {
        const float z = aux0[idx], hv = aux1[idx];
        v = (1.0f - z) * hv + z * tanhf(v);                              // new h
      }
      if (out32) out32[idx] = v;
      if (out16) out16[idx] = (_Float16)v;
    }
  }
}

// ---- Pack weight [K,H] f32 -> WMMA B-swizzled f16, zero padded -------------
__global__ void pack_b_kernel(const float* __restrict__ W, _Float16* __restrict__ Bp,
                              int K, int kChunks) {
  const int total = kChunks * CT * 512;
  int t = blockIdx.x * blockDim.x + threadIdx.x;
  if (t >= total) return;
  const int j    = t & 15;
  const int rest = t >> 4;
  const int lane = rest & 31;
  const int blk  = rest >> 5;           // kc*CT + ct
  const int ct   = blk % CT;
  const int kc   = blk / CT;
  const int col  = ct * 16 + (lane & 15);
  const int kk   = kc * 32 + (j & 8) * 2 + (lane >> 4) * 8 + (j & 7);
  Bp[t] = (kk < K && col < H) ? (_Float16)W[(size_t)kk * H + col] : (_Float16)0.0f;
}

__global__ void pad_bias_kernel(const float* __restrict__ b, float* __restrict__ bp) {
  int c = threadIdx.x;
  if (c < Hp) bp[c] = (c < H) ? b[c] : 0.0f;
}

__global__ void fill_kernel(float* __restrict__ p, float v, size_t n) {
  size_t i = (size_t)blockIdx.x * blockDim.x + threadIdx.x;
  if (i < n) p[i] = v;
}

__global__ void cvt_f32_f16_kernel(const float* __restrict__ in,
                                   _Float16* __restrict__ out, size_t n) {
  size_t i = (size_t)blockIdx.x * blockDim.x + threadIdx.x;
  if (i < n) out[i] = (_Float16)in[i];
}

// ---- Fold tower einsum into W_mix ------------------------------------------
// Wcomb[t*DT+d, j] = sum_e W_tower[t,d,e] * W_mix[t*DT+e, j]   (f32 [H,H])
__global__ void wcomb_kernel(const float* __restrict__ W_tower,
                             const float* __restrict__ W_mix,
                             float* __restrict__ Wcomb) {
  int t = blockIdx.x * blockDim.x + threadIdx.x;
  if (t >= H * H) return;
  int i = t / H, j = t - i * H;
  int tw = i / DT, d = i - tw * DT;
  float s = 0.0f;
#pragma unroll
  for (int e = 0; e < DT; ++e)
    s += W_tower[((size_t)tw * DT + d) * DT + e] * W_mix[((size_t)tw * DT + e) * H + j];
  Wcomb[t] = s;
}

// ---- Edge message + scatter (gate recomputed in-register) ------------------
__global__ void msg_kernel(const float* __restrict__ h,
                           const float* __restrict__ ef,
                           const float* __restrict__ W_edge,
                           const float* __restrict__ b_edge,
                           const int* __restrict__ src,
                           const int* __restrict__ dst,
                           float* __restrict__ agg) {
  size_t t = (size_t)blockIdx.x * blockDim.x + threadIdx.x;
  if (t >= (size_t)NE * H) return;
  int    c = (int)(t % H);
  size_t e = t / H;
  int s = src[e], d = dst[e];
  float g = b_edge[c];
#pragma unroll
  for (int k = 0; k < FE; ++k) g += ef[e * FE + k] * W_edge[(size_t)k * H + c];
  g = sigmoidf_(g);
  atomicAdd(&agg[(size_t)d * Hp + c], g * h[(size_t)s * Hp + c]);
}

// ---- Readout ---------------------------------------------------------------
DEVINL void atomicMaxF(float* addr, float val) {
  unsigned int* ua = (unsigned int*)addr;
  unsigned int old = *ua;
  while (__uint_as_float(old) < val) {
    unsigned int assumed = old;
    old = atomicCAS(ua, assumed, __float_as_uint(val));
    if (old == assumed) break;
  }
}

__global__ void score_kernel(const float* __restrict__ h,
                             const float* __restrict__ W_score,
                             const float* __restrict__ b_score,
                             const int* __restrict__ batch,
                             float* __restrict__ score, float* __restrict__ smax) {
  int n = blockIdx.x * blockDim.x + threadIdx.x;
  if (n >= NN) return;
  float s = b_score[0];
  for (int c = 0; c < H; ++c) s += h[(size_t)n * Hp + c] * W_score[c];
  score[n] = s;
  atomicMaxF(&smax[batch[n]], s);
}

__global__ void exden_kernel(const float* __restrict__ score,
                             const float* __restrict__ smax,
                             const int* __restrict__ batch,
                             float* __restrict__ ex, float* __restrict__ denom) {
  int n = blockIdx.x * blockDim.x + threadIdx.x;
  if (n >= NN) return;
  int g = batch[n];
  float e = __expf(score[n] - smax[g]);
  ex[n] = e;
  atomicAdd(&denom[g], e);
}

__global__ void attn_scatter_kernel(const float* __restrict__ ex,
                                    const float* __restrict__ denom,
                                    const int* __restrict__ batch,
                                    const float* __restrict__ feat,
                                    float* __restrict__ gsum) {
  size_t t = (size_t)blockIdx.x * blockDim.x + threadIdx.x;
  if (t >= (size_t)NN * H) return;
  int    c = (int)(t % H);
  size_t n = t / H;
  int g = batch[n];
  float alpha = ex[n] / denom[g];
  atomicAdd(&gsum[(size_t)g * Hp + c], alpha * feat[(size_t)n * Hp + c]);
}

__global__ void final_out_kernel(const float* __restrict__ hidden,
                                 const float* __restrict__ W_out,
                                 const float* __restrict__ b_out,
                                 float* __restrict__ out) {
  int g = blockIdx.x * blockDim.x + threadIdx.x;
  if (g >= NG) return;
  float s = b_out[0];
  for (int c = 0; c < H; ++c) s += hidden[(size_t)g * Hp + c] * W_out[c];
  out[g] = s;
}

// ---------------------------------------------------------------------------
static inline int gemm_blocks(int rowTiles) { return (rowTiles + 7) / 8; }

extern "C" void kernel_launch(void* const* d_in, const int* in_sizes, int n_in,
                              void* d_out, int out_size, void* d_ws, size_t ws_size,
                              hipStream_t stream) {
  (void)in_sizes; (void)n_in; (void)out_size; (void)ws_size;

  const float* node_features = (const float*)d_in[0];
  const float* edge_features = (const float*)d_in[1];
  const int*   edge_index    = (const int*)d_in[2];
  const int*   batch         = (const int*)d_in[3];
  const float* W_in   = (const float*)d_in[4];
  const float* b_in   = (const float*)d_in[5];
  const float* W_edge = (const float*)d_in[6];
  const float* b_edge = (const float*)d_in[7];
  const float* W_tower= (const float*)d_in[8];
  const float* W_mix  = (const float*)d_in[9];
  const float* b_mix  = (const float*)d_in[10];
  const float* Wz = (const float*)d_in[11];
  const float* Uz = (const float*)d_in[12];
  const float* bz = (const float*)d_in[13];
  const float* Wr = (const float*)d_in[14];
  const float* Ur = (const float*)d_in[15];
  const float* br = (const float*)d_in[16];
  const float* Wh = (const float*)d_in[17];
  const float* Uh = (const float*)d_in[18];
  const float* bh = (const float*)d_in[19];
  const float* W_score = (const float*)d_in[20];
  const float* b_score = (const float*)d_in[21];
  const float* W_feat  = (const float*)d_in[22];
  const float* b_feat  = (const float*)d_in[23];
  const float* W_r1    = (const float*)d_in[24];
  const float* b_r1    = (const float*)d_in[25];
  const float* W_out   = (const float*)d_in[26];
  const float* b_out   = (const float*)d_in[27];
  float* out = (float*)d_out;

  const int* src = edge_index;
  const int* dst = edge_index + NE;

  // ---- workspace carve ----
  const size_t NpHp = (size_t)NN * Hp;      // 22,400,000
  float* f = (float*)d_ws;
  float* h32    = f; f += NpHp;
  float* agg32  = f; f += NpHp;             // also reused as feat
  float* z32    = f; f += NpHp;
  float* gsum32 = f; f += (size_t)NG * Hp;
  float* hidden = f; f += (size_t)NG * Hp;
  float* Wcomb  = f; f += (size_t)H * H;
  float* score  = f; f += NN;
  float* ex     = f; f += NN;
  float* smax   = f; f += NG;
  float* denom  = f; f += NG;
  float* biasp  = f; f += 7 * Hp;
  float* bp_in   = biasp + 0 * Hp;
  float* bp_mix  = biasp + 1 * Hp;
  float* bp_z    = biasp + 2 * Hp;
  float* bp_r    = biasp + 3 * Hp;
  float* bp_h    = biasp + 4 * Hp;
  float* bp_feat = biasp + 5 * Hp;
  float* bp_r1   = biasp + 6 * Hp;

  _Float16* x = (_Float16*)f;
  _Float16* h16    = x; x += NpHp;
  _Float16* agg16  = x; x += NpHp;
  _Float16* m16    = x; x += NpHp;
  _Float16* rh16   = x; x += NpHp;
  _Float16* nf16   = x; x += (size_t)NN * FN;
  _Float16* gsum16 = x; x += (size_t)NG * Hp;
  const size_t packH = (size_t)KCH * CT * 512;   // 50176 halves
  const size_t packF = (size_t)KCF * CT * 512;   // 28672 halves
  _Float16* WinP   = x; x += packF;
  _Float16* WcombP = x; x += packH;
  _Float16* WzP    = x; x += packH;
  _Float16* UzP    = x; x += packH;
  _Float16* WrP    = x; x += packH;
  _Float16* UrP    = x; x += packH;
  _Float16* WhP    = x; x += packH;
  _Float16* UhP    = x; x += packH;
  _Float16* WfeatP = x; x += packH;
  _Float16* Wr1P   = x; x += packH;

  // ---- one-time setup: padded biases, folded+packed weights, f16 inputs ----
  pad_bias_kernel<<<1, 256, 0, stream>>>(b_in,   bp_in);
  pad_bias_kernel<<<1, 256, 0, stream>>>(b_mix,  bp_mix);
  pad_bias_kernel<<<1, 256, 0, stream>>>(bz,     bp_z);
  pad_bias_kernel<<<1, 256, 0, stream>>>(br,     bp_r);
  pad_bias_kernel<<<1, 256, 0, stream>>>(bh,     bp_h);
  pad_bias_kernel<<<1, 256, 0, stream>>>(b_feat, bp_feat);
  pad_bias_kernel<<<1, 256, 0, stream>>>(b_r1,   bp_r1);

  wcomb_kernel<<<(H * H + 255) / 256, 256, 0, stream>>>(W_tower, W_mix, Wcomb);

  const int pbH = (int)((packH + 255) / 256);
  const int pbF = (int)((packF + 255) / 256);
  pack_b_kernel<<<pbF, 256, 0, stream>>>(W_in,   WinP,   FN, KCF);
  pack_b_kernel<<<pbH, 256, 0, stream>>>(Wcomb,  WcombP, H,  KCH);
  pack_b_kernel<<<pbH, 256, 0, stream>>>(Wz,     WzP,    H,  KCH);
  pack_b_kernel<<<pbH, 256, 0, stream>>>(Uz,     UzP,    H,  KCH);
  pack_b_kernel<<<pbH, 256, 0, stream>>>(Wr,     WrP,    H,  KCH);
  pack_b_kernel<<<pbH, 256, 0, stream>>>(Ur,     UrP,    H,  KCH);
  pack_b_kernel<<<pbH, 256, 0, stream>>>(Wh,     WhP,    H,  KCH);
  pack_b_kernel<<<pbH, 256, 0, stream>>>(Uh,     UhP,    H,  KCH);
  pack_b_kernel<<<pbH, 256, 0, stream>>>(W_feat, WfeatP, H,  KCH);
  pack_b_kernel<<<pbH, 256, 0, stream>>>(W_r1,   Wr1P,   H,  KCH);

  const size_t nfN = (size_t)NN * FN;
  cvt_f32_f16_kernel<<<(int)((nfN + 255) / 256), 256, 0, stream>>>(node_features, nf16, nfN);

  // ---- input projection: h = relu(nf @ W_in + b_in) ----
  gemm_wmma_kernel<EPI_RELU, false><<<gemm_blocks(NN / 16), 256, 0, stream>>>(
      nf16, WinP, nullptr, nullptr, bp_in, nullptr, nullptr, h32, h16,
      NN / 16, KCF, FN);

  const int fillNpHp = (int)((NpHp + 255) / 256);
  const int msgBlks  = (int)(((size_t)NE * H + 255) / 256);

  for (int step = 0; step < STEPS; ++step) {
    fill_kernel<<<fillNpHp, 256, 0, stream>>>(agg32, 0.0f, NpHp);
    msg_kernel<<<msgBlks, 256, 0, stream>>>(h32, edge_features, W_edge, b_edge,
                                            src, dst, agg32);
    cvt_f32_f16_kernel<<<fillNpHp, 256, 0, stream>>>(agg32, agg16, NpHp);
    // m = relu(agg @ Wcomb + b_mix)  (f16 only; m is GEMM-A-only)
    gemm_wmma_kernel<EPI_RELU, false><<<gemm_blocks(NN / 16), 256, 0, stream>>>(
        agg16, WcombP, nullptr, nullptr, bp_mix, nullptr, nullptr,
        nullptr, m16, NN / 16, KCH, Hp);
    // z = sigmoid(m@Wz + h@Uz + bz)  (f32 only; aux use)
    gemm_wmma_kernel<EPI_SIG, true><<<gemm_blocks(NN / 16), 256, 0, stream>>>(
        m16, WzP, h16, UzP, bp_z, nullptr, nullptr,
        z32, nullptr, NN / 16, KCH, Hp);
    // rh = sigmoid(m@Wr + h@Ur + br) * h  (f16 only; GEMM-A-only)
    gemm_wmma_kernel<EPI_SIGMUL, true><<<gemm_blocks(NN / 16), 256, 0, stream>>>(
        m16, WrP, h16, UrP, bp_r, h32, nullptr,
        nullptr, rh16, NN / 16, KCH, Hp);
    // h = (1-z)*h + z*tanh(m@Wh + rh@Uh + bh)  (dual-store; in-place on h)
    gemm_wmma_kernel<EPI_GRU, true><<<gemm_blocks(NN / 16), 256, 0, stream>>>(
        m16, WhP, rh16, UhP, bp_h, z32, h32,
        h32, h16, NN / 16, KCH, Hp);
  }

  // ---- attention readout ----
  fill_kernel<<<1, 256, 0, stream>>>(smax, -3.0e38f, (size_t)NG);
  fill_kernel<<<1, 256, 0, stream>>>(denom, 0.0f, (size_t)NG);
  fill_kernel<<<(NG * Hp + 255) / 256, 256, 0, stream>>>(gsum32, 0.0f, (size_t)NG * Hp);

  score_kernel<<<(NN + 255) / 256, 256, 0, stream>>>(h32, W_score, b_score, batch, score, smax);
  exden_kernel<<<(NN + 255) / 256, 256, 0, stream>>>(score, smax, batch, ex, denom);

  // feat = h @ W_feat + b_feat   (f32 only, into agg32)
  gemm_wmma_kernel<EPI_NONE, false><<<gemm_blocks(NN / 16), 256, 0, stream>>>(
      h16, WfeatP, nullptr, nullptr, bp_feat, nullptr, nullptr,
      agg32, nullptr, NN / 16, KCH, Hp);
  attn_scatter_kernel<<<fillNpHp, 256, 0, stream>>>(ex, denom, batch, agg32, gsum32);

  cvt_f32_f16_kernel<<<(NG * Hp + 255) / 256, 256, 0, stream>>>(gsum32, gsum16, (size_t)NG * Hp);
  // hidden = relu(gsum @ W_r1 + b_r1)
  gemm_wmma_kernel<EPI_RELU, false><<<gemm_blocks(NG / 16), 256, 0, stream>>>(
      gsum16, Wr1P, nullptr, nullptr, bp_r1, nullptr, nullptr,
      hidden, nullptr, NG / 16, KCH, Hp);
  final_out_kernel<<<1, NG, 0, stream>>>(hidden, W_out, b_out, out);
}
